// FrequencyDomainMLP_18966575579743
// MI455X (gfx1250) — compile-verified, matrix operand order checked by reading, not run on hardware
//
#include <hip/hip_runtime.h>
#include <cstdint>
#include <cstddef>

#define B_    256
#define SEQ_  200
#define H_    256
#define NFFT_ 16
#define HOP_  8
#define FREQ_ 9
#define NSEG_ 26
#define FACT_ 2

typedef __attribute__((ext_vector_type(16))) __bf16 v16bf;
typedef __attribute__((ext_vector_type(8)))  float  v8f;

// ---- CDNA5 async global->LDS (ASYNCcnt) with safe fallback --------------
#if defined(__has_builtin)
#if __has_builtin(__builtin_amdgcn_load_to_lds)
#define HAS_ASYNC_LDS 1
#endif
#endif
#ifndef HAS_ASYNC_LDS
#define HAS_ASYNC_LDS 0
#endif

__device__ __forceinline__ void cp_g2l_b32(const void* g, void* l) {
#if HAS_ASYNC_LDS
  __builtin_amdgcn_load_to_lds(const_cast<void*>(g),
                               (__attribute__((address_space(3))) void*)l,
                               /*size=*/4, /*offset=*/0, /*aux=*/0);
#else
  *(uint32_t*)l = *(const uint32_t*)g;
#endif
}
__device__ __forceinline__ void async_fence() {
#if HAS_ASYNC_LDS
  asm volatile("s_wait_asynccnt 0" ::: "memory");
#endif
}

// cos(k*pi/8), sin(k*pi/8), k = 0..15  (NFFT = 16 twiddles)
__constant__ float C16[16] = {
  1.0f, 0.92387953251128674f, 0.70710678118654757f, 0.38268343236508978f,
  0.0f,-0.38268343236508978f,-0.70710678118654757f,-0.92387953251128674f,
 -1.0f,-0.92387953251128674f,-0.70710678118654757f,-0.38268343236508978f,
  0.0f, 0.38268343236508978f, 0.70710678118654757f, 0.92387953251128674f };
__constant__ float S16[16] = {
  0.0f, 0.38268343236508978f, 0.70710678118654757f, 0.92387953251128674f,
  1.0f, 0.92387953251128674f, 0.70710678118654757f, 0.38268343236508978f,
  0.0f,-0.38268343236508978f,-0.70710678118654757f,-0.92387953251128674f,
 -1.0f,-0.92387953251128674f,-0.70710678118654757f,-0.38268343236508978f };

__device__ __forceinline__ float gelu_f(float x) {
  return 0.5f * x * (1.0f + erff(x * 0.70710678118654752f));
}
__device__ __forceinline__ __bf16 f2bf(float x) { return (__bf16)x; }
__device__ __forceinline__ float  ld_f(const float*  p, long i) { return p[i]; }
__device__ __forceinline__ float  ld_f(const __bf16* p, long i) { return (float)p[i]; }
__device__ __forceinline__ void   st_f(float*  p, long i, float v) { p[i] = v; }
__device__ __forceinline__ void   st_f(__bf16* p, long i, float v) { p[i] = (__bf16)v; }

union FragU { v16bf v; uint4 q[2]; };

// 16x32 bf16 WMMA operand fragment from LDS (ISA 7.12.2 layout):
// lane<16 holds k-chunks [0..7],[16..23] of row (lane&15); lane>=16 holds
// [8..15],[24..31]. Two 16B ds_load_b128 per fragment.
__device__ __forceinline__ v16bf frag_ld(const __bf16* base, int row, int stride, int k0) {
  FragU f;
  const __bf16* p = base + (long)row * stride + k0;
  f.q[0] = *(const uint4*)(p);
  f.q[1] = *(const uint4*)(p + 16);
  return f.v;
}

__device__ __forceinline__ v8f wmma_bf(v16bf a, v16bf b, v8f c) {
  return __builtin_amdgcn_wmma_f32_16x16x32_bf16(false, a, false, b, (short)0, c, false, false);
}

// ---------------------------------------------------------------------------
// 1) gate init:  w[b,s] = conv_b[s] + exp(s + 1 - NSEG)
// ---------------------------------------------------------------------------
__global__ void init_w_kernel(const float* __restrict__ conv_b, float* __restrict__ wbuf) {
  int i = blockIdx.x * blockDim.x + threadIdx.x;
  if (i < B_ * NSEG_) {
    int s = i % NSEG_;
    wbuf[i] = conv_b[s] + expf((float)(s + 1 - NSEG_));
  }
}

// ---------------------------------------------------------------------------
// 2) STFT (16-pt rDFT) + fused |X| . conv_w reduction (atomic into wbuf)
//    spec layout: xr/xi [b][h][f][s]  (s fastest)
// ---------------------------------------------------------------------------
__global__ void stft_kernel(const float* __restrict__ x, const float* __restrict__ conv_w,
                            float* __restrict__ xr, float* __restrict__ xi,
                            float* __restrict__ wbuf) {
  long gid = (long)blockIdx.x * blockDim.x + threadIdx.x;
  if (gid >= (long)B_ * H_ * NSEG_) return;
  int s = (int)(gid % NSEG_);
  long r = gid / NSEG_;
  int h = (int)(r % H_);
  int b = (int)(r / H_);

  float fr[NFFT_];
#pragma unroll
  for (int i = 0; i < NFFT_; i++) {
    int t = s * HOP_ - NFFT_ / 2 + i;
    fr[i] = (t >= 0 && t < SEQ_) ? x[((long)b * SEQ_ + t) * H_ + h] : 0.0f;
  }
  const float scale = 0.25f;  // NFFT^-0.5
  long base = ((long)(b * H_ + h)) * FREQ_ * NSEG_ + s;
  float part = 0.0f;
#pragma unroll
  for (int f = 0; f < FREQ_; f++) {
    float vr = 0.0f, vi = 0.0f;
#pragma unroll
    for (int i = 0; i < NFFT_; i++) {
      int k = (f * i) & 15;
      vr += fr[i] * C16[k];
      vi -= fr[i] * S16[k];
    }
    vr *= scale; vi *= scale;
    xr[base + (long)f * NSEG_] = vr;
    xi[base + (long)f * NSEG_] = vi;
    part += sqrtf(vr * vr + vi * vi) * conv_w[((long)s * H_ + h) * FREQ_ + f];
  }
  atomicAdd(&wbuf[b * NSEG_ + s], part);
}

// ---------------------------------------------------------------------------
// 3) apply gate
// ---------------------------------------------------------------------------
__global__ void scale_kernel(float* __restrict__ xr, float* __restrict__ xi,
                             const float* __restrict__ wbuf) {
  long gid = (long)blockIdx.x * blockDim.x + threadIdx.x;
  if (gid >= (long)B_ * H_ * FREQ_ * NSEG_) return;
  int s = (int)(gid % NSEG_);
  int b = (int)(gid / ((long)H_ * FREQ_ * NSEG_));
  float w = wbuf[b * NSEG_ + s];
  xr[gid] *= w;
  xi[gid] *= w;
}

// ---------------------------------------------------------------------------
// 4) fused small complex mixer (freq: 9->18->9, seg: 26->52->26)
//    Both layers fully in LDS, zero-padded to WMMA tiles. In-place on xr/xi
//    (each block's read/write footprint is identical & disjoint).
//    Addressing: elem(j,m,k) = m*am + j*ajs + k*aks  (output n uses aks).
// ---------------------------------------------------------------------------
__global__ void __launch_bounds__(256) small_mixer_kernel(
    float* __restrict__ Xr, float* __restrict__ Xi,
    const float* __restrict__ W10, const float* __restrict__ W11,
    const float* __restrict__ b10, const float* __restrict__ b11,
    const float* __restrict__ W20, const float* __restrict__ W21,
    const float* __restrict__ b20, const float* __restrict__ b21,
    int M, int K1, int N1, long am, long ajs, long aks) {
  __shared__ __align__(16) __bf16 sAr[64][32], sAi[64][32];
  __shared__ __align__(16) __bf16 sW10[64][32], sW11[64][32];  // [n][k] transposed
  __shared__ __align__(16) __bf16 sW20[32][64], sW21[32][64];  // [n][k] transposed
  __shared__ __align__(16) __bf16 sO1r[64][64], sO1i[64][64];

  int j = blockIdx.y;
  int m0 = blockIdx.x * 64;
  int tid = threadIdx.x, lane = tid & 31, wave = tid >> 5;
  int lrow = lane & 15, lk0 = (lane & 16) ? 8 : 0;

#pragma unroll
  for (int u = 0; u < 8; u++) {
    int idx = tid + u * 256;  // 0..2047
    // A tile (64 x 32)
    int row = idx >> 5, kk = idx & 31;
    int m = m0 + row;
    float vr = 0.0f, vi = 0.0f;
    if (m < M && kk < K1) {
      long a = (long)m * am + (long)j * ajs + (long)kk * aks;
      vr = Xr[a]; vi = Xi[a];
    }
    sAr[row][kk] = f2bf(vr); sAi[row][kk] = f2bf(vi);
    // W1 (K1 x N1), padded to 32x64, transposed into [n][k]
    int n1 = idx >> 5, k1 = idx & 31;
    float w0 = 0.0f, w1 = 0.0f;
    if (k1 < K1 && n1 < N1) {
      long a = ((long)j * K1 + k1) * N1 + n1;
      w0 = W10[a]; w1 = W11[a];
    }
    sW10[n1][k1] = f2bf(w0); sW11[n1][k1] = f2bf(w1);
    // W2 (N1 x K1), padded to 64x32, transposed into [n][k]
    int n2 = idx >> 6, k2 = idx & 63;
    float u0 = 0.0f, u1 = 0.0f;
    if (k2 < N1 && n2 < K1) {
      long a = ((long)j * N1 + k2) * K1 + n2;
      u0 = W20[a]; u1 = W21[a];
    }
    sW20[n2][k2] = f2bf(u0); sW21[n2][k2] = f2bf(u1);
  }
  __syncthreads();

  // layer 1: 4 m-tiles x 4 n-tiles = 16 jobs, 8 waves x 2
  for (int jb = wave; jb < 16; jb += 8) {
    int tm = jb >> 2, tn = jb & 3;
    v16bf ar = frag_ld(&sAr[0][0], tm * 16 + lrow, 32, lk0);
    v16bf ai = frag_ld(&sAi[0][0], tm * 16 + lrow, 32, lk0);
    v16bf w0 = frag_ld(&sW10[0][0], tn * 16 + lrow, 32, lk0);
    v16bf w1 = frag_ld(&sW11[0][0], tn * 16 + lrow, 32, lk0);
    v8f aA = {}, aB = {}, aC = {}, aD = {};
    aA = wmma_bf(ar, w0, aA);
    aB = wmma_bf(ai, w1, aB);
    aC = wmma_bf(ai, w0, aC);
    aD = wmma_bf(ar, w1, aD);
#pragma unroll
    for (int i = 0; i < 8; i++) {
      int ml = tm * 16 + i + ((lane >> 4) << 3);
      int nl = tn * 16 + lrow;
      float vr = 0.0f, vi = 0.0f;
      if (nl < N1) {
        float bb0 = b10[(long)j * N1 + nl], bb1 = b11[(long)j * N1 + nl];
        vr = gelu_f(aA[i] - aB[i] + bb0);
        vi = gelu_f(aC[i] + aD[i] + bb1);
      }
      sO1r[ml][nl] = f2bf(vr);
      sO1i[ml][nl] = f2bf(vi);
    }
  }
  __syncthreads();

  // layer 2 (quirk: o2i = o1i@(W20+W21)+b): 4 m-tiles x 2 n-tiles, 1 per wave
  {
    int tm = wave >> 1, tn = wave & 1;
    v8f aA = {}, aB = {}, aC = {};
#pragma unroll
    for (int kt = 0; kt < 2; kt++) {
      int kb = kt * 32 + lk0;
      v16bf orr = frag_ld(&sO1r[0][0], tm * 16 + lrow, 64, kb);
      v16bf oii = frag_ld(&sO1i[0][0], tm * 16 + lrow, 64, kb);
      v16bf w0 = frag_ld(&sW20[0][0], tn * 16 + lrow, 64, kb);
      v16bf w1 = frag_ld(&sW21[0][0], tn * 16 + lrow, 64, kb);
      aA = wmma_bf(orr, w0, aA);
      aB = wmma_bf(oii, w1, aB);
      aC = wmma_bf(oii, w0, aC);
    }
#pragma unroll
    for (int i = 0; i < 8; i++) {
      int ml = tm * 16 + i + ((lane >> 4) << 3);
      int nl = tn * 16 + lrow;
      int m = m0 + ml;
      if (m < M && nl < K1) {
        float bb0 = b20[(long)j * K1 + nl], bb1 = b21[(long)j * K1 + nl];
        float vr = aA[i] - aB[i] + bb0;
        float vi = aC[i] + aB[i] + bb1;
        long o = (long)m * am + (long)j * ajs + (long)nl * aks;
        Xr[o] = vr;
        Xi[o] = vi;
      }
    }
  }
}

// ---------------------------------------------------------------------------
// 5) generic complex-mixer GEMM layer (channel mixer, big K)
//    BM=64 x BN=64 block tile, 8 waves x 2 C-tile jobs (shared W fragment).
//    A addr(j,m,k) = (m/adiv)*aos1 + (m%adiv)*aos2 + j*ajs + k*aks
//    O addr(j,m,n) = (m/odiv)*oos1 + (m%odiv)*oos2 + j*ojs + n*ons
//    ACONT: A is contiguous bf16 rows (addr = j*ajs + m*aos1 + k); uses the
//    CDNA5 async global->LDS path. Requires M%64==0 && K%32==0.
// ---------------------------------------------------------------------------
template <bool GELU, bool QUIRK, bool ACONT, typename AT, typename OT>
__global__ void __launch_bounds__(256) cplx_gemm_kernel(
    const AT* __restrict__ Ar, const AT* __restrict__ Ai,
    const float* __restrict__ W0, const float* __restrict__ W1,
    const float* __restrict__ b0, const float* __restrict__ b1,
    OT* __restrict__ Or, OT* __restrict__ Oi,
    int M, int K, int N,
    int adiv, long aos1, long aos2, long ajs, long aks,
    int odiv, long oos1, long oos2, long ojs, long ons) {
  const int BM = 64, BN = 64, BK = 32;
  __shared__ __align__(16) __bf16 sAr[BM][BK], sAi[BM][BK];
  __shared__ __align__(16) __bf16 sW0[BN][BK], sW1[BN][BK];  // [n][k]

  int j = blockIdx.z;
  int m0 = blockIdx.x * BM, n0 = blockIdx.y * BN;
  int tid = threadIdx.x, lane = tid & 31, wave = tid >> 5;
  int tn = wave & 3, tmA = wave >> 2;  // jobs: (tmA, tn), (tmA+2, tn)
  int lrow = lane & 15, lk0 = (lane & 16) ? 8 : 0;

  v8f aA[2] = {}, aB[2] = {}, aC[2] = {}, aD[2] = {};
  int nkt = (K + BK - 1) / BK;

  for (int kt = 0; kt < nkt; kt++) {
    int kg0 = kt * BK;
    if (kt + 1 < nkt)  // speculative prefetch of next W tile
      __builtin_prefetch(&W0[((long)j * K + kg0 + BK) * N + n0], 0, 1);
    if (ACONT) {
      // contiguous bf16 rows: copy 4B words straight into LDS (async on CDNA5)
#pragma unroll
      for (int u = 0; u < 4; u++) {
        int word = tid + u * 256;      // 1024 words = 64 rows x 16 words
        int row = word >> 4;
        int kw = (word & 15) << 1;     // bf16 k within tile
        long g = (long)j * ajs + (long)(m0 + row) * aos1 + (kg0 + kw);
        cp_g2l_b32((const void*)(Ar + g), (void*)&sAr[row][kw]);
        cp_g2l_b32((const void*)(Ai + g), (void*)&sAi[row][kw]);
      }
    } else {
#pragma unroll
      for (int u = 0; u < 8; u++) {  // A: 64x32 = 2048 elems
        int idx = tid + u * 256;
        int row = idx >> 5, kk = idx & 31;
        int m = m0 + row, gk = kg0 + kk;
        float vr = 0.0f, vi = 0.0f;
        if (m < M && gk < K) {
          long a = (long)(m / adiv) * aos1 + (long)(m % adiv) * aos2 +
                   (long)j * ajs + (long)gk * aks;
          vr = ld_f(Ar, a);
          vi = ld_f(Ai, a);
        }
        sAr[row][kk] = f2bf(vr);
        sAi[row][kk] = f2bf(vi);
      }
    }
#pragma unroll
    for (int u = 0; u < 8; u++) {  // W: 32x64 = 2048 elems (transposed into LDS)
      int idx = tid + u * 256;
      int nn = idx >> 5, kk = idx & 31;
      int gk = kg0 + kk, n = n0 + nn;
      float w0 = 0.0f, w1 = 0.0f;
      if (gk < K && n < N) {
        long a = ((long)j * K + gk) * N + n;
        w0 = W0[a];
        w1 = W1[a];
      }
      sW0[nn][kk] = f2bf(w0);
      sW1[nn][kk] = f2bf(w1);
    }
    if (ACONT) async_fence();
    __syncthreads();
    v16bf w0f = frag_ld(&sW0[0][0], tn * 16 + lrow, BK, lk0);
    v16bf w1f = frag_ld(&sW1[0][0], tn * 16 + lrow, BK, lk0);
#pragma unroll
    for (int jj = 0; jj < 2; jj++) {
      int tm = tmA + jj * 2;
      v16bf ar = frag_ld(&sAr[0][0], tm * 16 + lrow, BK, lk0);
      v16bf ai = frag_ld(&sAi[0][0], tm * 16 + lrow, BK, lk0);
      aA[jj] = wmma_bf(ar, w0f, aA[jj]);
      aB[jj] = wmma_bf(ai, w1f, aB[jj]);
      aC[jj] = wmma_bf(ai, w0f, aC[jj]);
      if (!QUIRK) aD[jj] = wmma_bf(ar, w1f, aD[jj]);
    }
    __syncthreads();
  }

#pragma unroll
  for (int jj = 0; jj < 2; jj++) {
    int tm = tmA + jj * 2;
#pragma unroll
    for (int i = 0; i < 8; i++) {
      int m = m0 + tm * 16 + i + ((lane >> 4) << 3);
      int n = n0 + tn * 16 + lrow;
      if (m < M && n < N) {
        float bb0 = b0[(long)j * N + n], bb1 = b1[(long)j * N + n];
        float vr = aA[jj][i] - aB[jj][i] + bb0;
        float vi = (QUIRK ? (aC[jj][i] + aB[jj][i]) : (aC[jj][i] + aD[jj][i])) + bb1;
        if (GELU) { vr = gelu_f(vr); vi = gelu_f(vi); }
        long o = (long)(m / odiv) * oos1 + (long)(m % odiv) * oos2 +
                 (long)j * ojs + (long)n * ons;
        st_f(Or, o, vr);
        st_f(Oi, o, vi);
      }
    }
  }
}

// ---------------------------------------------------------------------------
// 6) plain GEMM for FF block: Out = op(A@W + b) [+ Res] ; A (M,K), W (K,N)
//    BM=64 x BN=64 tile, 2 C-tile jobs per wave. ACONT as above (bf16 A).
// ---------------------------------------------------------------------------
template <bool GELU, bool RES, bool ACONT, typename AT, typename OT>
__global__ void __launch_bounds__(256) gemm_kernel(
    const AT* __restrict__ A, const float* __restrict__ W, const float* __restrict__ bias,
    const float* __restrict__ Res, OT* __restrict__ Out, int M, int K, int N) {
  const int BM = 64, BN = 64, BK = 32;
  __shared__ __align__(16) __bf16 sA[BM][BK];
  __shared__ __align__(16) __bf16 sW[BN][BK];

  int m0 = blockIdx.x * BM, n0 = blockIdx.y * BN;
  int tid = threadIdx.x, lane = tid & 31, wave = tid >> 5;
  int tn = wave & 3, tmA = wave >> 2;
  int lrow = lane & 15, lk0 = (lane & 16) ? 8 : 0;

  v8f acc[2] = {};
  int nkt = (K + BK - 1) / BK;
  for (int kt = 0; kt < nkt; kt++) {
    int kg0 = kt * BK;
    if (kt + 1 < nkt)
      __builtin_prefetch(&W[(long)(kg0 + BK) * N + n0], 0, 1);
    if (ACONT) {
#pragma unroll
      for (int u = 0; u < 4; u++) {
        int word = tid + u * 256;
        int row = word >> 4;
        int kw = (word & 15) << 1;
        long g = (long)(m0 + row) * K + (kg0 + kw);
        cp_g2l_b32((const void*)(A + g), (void*)&sA[row][kw]);
      }
    } else {
#pragma unroll
      for (int u = 0; u < 8; u++) {
        int idx = tid + u * 256;
        int row = idx >> 5, kk = idx & 31;
        int m = m0 + row, gk = kg0 + kk;
        float v = 0.0f;
        if (m < M && gk < K) v = ld_f(A, (long)m * K + gk);
        sA[row][kk] = f2bf(v);
      }
    }
#pragma unroll
    for (int u = 0; u < 8; u++) {
      int idx = tid + u * 256;
      int nn = idx >> 5, kk = idx & 31;
      int gk = kg0 + kk, n = n0 + nn;
      float v = 0.0f;
      if (gk < K && n < N) v = W[(long)gk * N + n];
      sW[nn][kk] = f2bf(v);
    }
    if (ACONT) async_fence();
    __syncthreads();
    v16bf wf = frag_ld(&sW[0][0], tn * 16 + lrow, BK, lk0);
#pragma unroll
    for (int jj = 0; jj < 2; jj++) {
      int tm = tmA + jj * 2;
      v16bf af = frag_ld(&sA[0][0], tm * 16 + lrow, BK, lk0);
      acc[jj] = wmma_bf(af, wf, acc[jj]);
    }
    __syncthreads();
  }

#pragma unroll
  for (int jj = 0; jj < 2; jj++) {
    int tm = tmA + jj * 2;
#pragma unroll
    for (int i = 0; i < 8; i++) {
      int m = m0 + tm * 16 + i + ((lane >> 4) << 3);
      int n = n0 + tn * 16 + lrow;
      if (m < M && n < N) {
        float v = acc[jj][i] + bias[n];
        if (GELU) v = gelu_f(v);
        if (RES) v += Res[(long)m * N + n];
        st_f(Out, (long)m * N + n, v);
      }
    }
  }
}

// ---------------------------------------------------------------------------
// 7) ISTFT (overlap-add with env division) + residual:  z[b][t][h] = y + input
// ---------------------------------------------------------------------------
__global__ void istft_add_kernel(const float* __restrict__ xr, const float* __restrict__ xi,
                                 const float* __restrict__ inp, float* __restrict__ z) {
  long gid = (long)blockIdx.x * blockDim.x + threadIdx.x;
  if (gid >= (long)B_ * H_ * SEQ_) return;
  int t = (int)(gid % SEQ_);
  long r = gid / SEQ_;
  int h = (int)(r % H_);
  int b = (int)(r / H_);

  int p = t + NFFT_ / 2;
  int smin = (p >= 15) ? ((p - 8) >> 3) : 0;  // ceil((p-15)/8)
  int smax = p >> 3;
  if (smax > NSEG_ - 1) smax = NSEG_ - 1;

  long base = ((long)(b * H_ + h)) * FREQ_ * NSEG_;
  float acc = 0.0f;
  int cnt = 0;
  for (int s = smin; s <= smax; s++) {
    int i = p - s * HOP_;
    float v = xr[base + s];                                     // f = 0
    v += ((i & 1) ? -1.0f : 1.0f) * xr[base + 8L * NSEG_ + s];  // f = 8
#pragma unroll
    for (int f = 1; f < 8; f++) {
      int k = (f * i) & 15;
      v += 2.0f * (xr[base + (long)f * NSEG_ + s] * C16[k] -
                   xi[base + (long)f * NSEG_ + s] * S16[k]);
    }
    acc += 0.25f * v;  // (1/NFFT) * NFFT^0.5
    cnt++;
  }
  long o = ((long)b * SEQ_ + t) * H_ + h;
  z[o] = acc / (float)cnt + inp[o];
}

// ---------------------------------------------------------------------------
// 8) layernorm over H (=256 = blockDim), eps = 1e-12
// ---------------------------------------------------------------------------
__global__ void layernorm_kernel(const float* __restrict__ x, const float* __restrict__ g,
                                 const float* __restrict__ be, float* __restrict__ out) {
  __shared__ float red[H_];
  int tid = threadIdx.x;
  long base = (long)blockIdx.x * H_;
  float v = x[base + tid];
  red[tid] = v;
  __syncthreads();
  for (int s = H_ / 2; s > 0; s >>= 1) {
    if (tid < s) red[tid] += red[tid + s];
    __syncthreads();
  }
  float mean = red[0] / (float)H_;
  __syncthreads();
  float d = v - mean;
  red[tid] = d * d;
  __syncthreads();
  for (int s = H_ / 2; s > 0; s >>= 1) {
    if (tid < s) red[tid] += red[tid + s];
    __syncthreads();
  }
  float var = red[0] / (float)H_;
  out[base + tid] = g[tid] * d / sqrtf(var + 1e-12f) + be[tid];
}

// ---------------------------------------------------------------------------
extern "C" void kernel_launch(void* const* d_in, const int* in_sizes, int n_in,
                              void* d_out, int out_size, void* d_ws, size_t ws_size,
                              hipStream_t stream) {
  const float* input  = (const float*)d_in[0];
  const float* conv_w = (const float*)d_in[1];
  const float* conv_b = (const float*)d_in[2];
  const float* fm_w1  = (const float*)d_in[3];
  const float* fm_b1  = (const float*)d_in[4];
  const float* fm_w2  = (const float*)d_in[5];
  const float* fm_b2  = (const float*)d_in[6];
  const float* cm_w1  = (const float*)d_in[7];
  const float* cm_b1  = (const float*)d_in[8];
  const float* cm_w2  = (const float*)d_in[9];
  const float* cm_b2  = (const float*)d_in[10];
  const float* sm_w1  = (const float*)d_in[11];
  const float* sm_b1  = (const float*)d_in[12];
  const float* sm_w2  = (const float*)d_in[13];
  const float* sm_b2  = (const float*)d_in[14];
  const float* ln_w   = (const float*)d_in[15];
  const float* ln_b   = (const float*)d_in[16];
  const float* ff_w1  = (const float*)d_in[17];
  const float* ff_b1  = (const float*)d_in[18];
  const float* ff_w2  = (const float*)d_in[19];
  const float* ff_b2  = (const float*)d_in[20];
  const float* ffln_w = (const float*)d_in[21];
  const float* ffln_b = (const float*)d_in[22];

  const long SPEC = (long)B_ * H_ * FREQ_ * NSEG_;   // 15,335,424
  const long BT   = (long)B_ * SEQ_;                 // 51,200
  const int  MCH  = B_ * FREQ_;                      // 2304 (channel-mixer rows)
  const long O1CH = (long)NSEG_ * MCH * (H_ * FACT_);

  char* ws = (char*)d_ws;
  auto alloc = [&](size_t bytes) -> char* {
    char* p = ws;
    ws += (bytes + 255) & ~(size_t)255;
    return p;
  };
  float*  xr    = (float*)alloc(SPEC * 4);
  float*  xi    = (float*)alloc(SPEC * 4);
  float*  wbuf  = (float*)alloc((size_t)B_ * NSEG_ * 4);
  __bf16* o1r   = (__bf16*)alloc(O1CH * 2);
  __bf16* o1i   = (__bf16*)alloc(O1CH * 2);
  float*  z     = (float*)alloc(BT * H_ * 4);        // reused: ln1 in, ff2 out
  float*  hs    = (float*)alloc(BT * H_ * 4);
  __bf16* ffmid = (__bf16*)alloc(BT * (4L * H_) * 2);

  // --- STFT + gate ---
  init_w_kernel<<<(B_ * NSEG_ + 255) / 256, 256, 0, stream>>>(conv_b, wbuf);
  stft_kernel<<<(int)(((long)B_ * H_ * NSEG_ + 255) / 256), 256, 0, stream>>>(
      input, conv_w, xr, xi, wbuf);
  scale_kernel<<<(int)((SPEC + 255) / 256), 256, 0, stream>>>(xr, xi, wbuf);

  // --- freq mixer: J=NSEG(s), M=B*H, K=FREQ, elem = m*234 + j*1 + k*26 ---
  small_mixer_kernel<<<dim3((B_ * H_) / 64, NSEG_), 256, 0, stream>>>(
      xr, xi,
      fm_w1, fm_w1 + (long)NSEG_ * FREQ_ * (FREQ_ * FACT_),
      fm_b1, fm_b1 + (long)NSEG_ * (FREQ_ * FACT_),
      fm_w2, fm_w2 + (long)NSEG_ * (FREQ_ * FACT_) * FREQ_,
      fm_b2, fm_b2 + (long)NSEG_ * FREQ_,
      B_ * H_, FREQ_, FREQ_ * FACT_,
      (long)FREQ_ * NSEG_, 1L, (long)NSEG_);

  // --- channel mixer: J=NSEG(s), M=(b,f)=2304, K=H
  //     A elem = (m/9)*(H*234) + (m%9)*26 + j*1 + k*234
  cplx_gemm_kernel<true, false, false, float, __bf16>
      <<<dim3(MCH / 64, (H_ * FACT_) / 64, NSEG_), 256, 0, stream>>>(
          xr, xi,
          cm_w1, cm_w1 + (long)NSEG_ * H_ * (H_ * FACT_),
          cm_b1, cm_b1 + (long)NSEG_ * (H_ * FACT_),
          o1r, o1i,
          MCH, H_, H_ * FACT_,
          FREQ_, (long)H_ * FREQ_ * NSEG_, (long)NSEG_, 1L, (long)FREQ_ * NSEG_,
          1, (long)(H_ * FACT_), 0L, (long)MCH * (H_ * FACT_), 1L);
  // layer 2: A = o1 (bf16, contiguous rows, K=512) -> async LDS path
  cplx_gemm_kernel<false, true, true, __bf16, float>
      <<<dim3(MCH / 64, H_ / 64, NSEG_), 256, 0, stream>>>(
          o1r, o1i,
          cm_w2, cm_w2 + (long)NSEG_ * (H_ * FACT_) * H_,
          cm_b2, cm_b2 + (long)NSEG_ * H_,
          xr, xi,
          MCH, H_ * FACT_, H_,
          1, (long)(H_ * FACT_), 0L, (long)MCH * (H_ * FACT_), 1L,
          FREQ_, (long)H_ * FREQ_ * NSEG_, (long)NSEG_, 1L, (long)FREQ_ * NSEG_);

  // --- seg mixer: J=FREQ(f), M=B*H, K=NSEG, elem = m*234 + j*26 + k*1 ---
  small_mixer_kernel<<<dim3((B_ * H_) / 64, FREQ_), 256, 0, stream>>>(
      xr, xi,
      sm_w1, sm_w1 + (long)FREQ_ * NSEG_ * (NSEG_ * FACT_),
      sm_b1, sm_b1 + (long)FREQ_ * (NSEG_ * FACT_),
      sm_w2, sm_w2 + (long)FREQ_ * (NSEG_ * FACT_) * NSEG_,
      sm_b2, sm_b2 + (long)FREQ_ * NSEG_,
      B_ * H_, NSEG_, NSEG_ * FACT_,
      (long)FREQ_ * NSEG_, (long)NSEG_, 1L);

  // --- ISTFT + residual, LN ---
  istft_add_kernel<<<(int)(((long)B_ * H_ * SEQ_ + 255) / 256), 256, 0, stream>>>(
      xr, xi, input, z);
  layernorm_kernel<<<(int)BT, H_, 0, stream>>>(z, ln_w, ln_b, hs);

  // --- FF block ---
  gemm_kernel<true, false, false, float, __bf16>
      <<<dim3((int)(BT / 64), (4 * H_) / 64), 256, 0, stream>>>(
          hs, ff_w1, ff_b1, nullptr, ffmid, (int)BT, H_, 4 * H_);
  // FF layer 2: A = ffmid (bf16 contiguous, K=1024) -> async LDS path
  gemm_kernel<false, true, true, __bf16, float>
      <<<dim3((int)(BT / 64), H_ / 64), 256, 0, stream>>>(
          ffmid, ff_w2, ff_b2, hs, z, (int)BT, 4 * H_, H_);
  layernorm_kernel<<<(int)BT, H_, 0, stream>>>(z, ffln_w, ffln_b, (float*)d_out);

  (void)in_sizes; (void)n_in; (void)out_size; (void)ws_size;
}
